// VectorQuantize_72361609003146
// MI455X (gfx1250) — matrix operand order repaired and checked by reading, not compiled, not run on hardware
//
#include <hip/hip_runtime.h>

typedef __attribute__((ext_vector_type(16))) __bf16 v16bf;
typedef __attribute__((ext_vector_type(8)))  __bf16 v8bf;
typedef __attribute__((ext_vector_type(8)))  float  v8f;

#define N_ROWS 32768
#define DIM    256
#define KCODES 4096
#define KC     64                        // codebook rows per LDS chunk
#define MTILE  256                       // 8 waves * 32 rows per workgroup
#define KHALF  (KCODES / 2)              // K-split across blockIdx.y
#define NCHUNK_H (KHALF / KC)            // 32 chunks per half
#define CHUNK_BYTES (KC * DIM * 2)       // 32 KB bf16 chunk

// ---- scratch layout carved from the out region (fully rewritten later) ----
#define SCR_CB    0u                                   // bf16 codebook, 2 MB
#define SCR_C2    (KCODES * DIM * 2u)                  // fp32 norms, 16 KB
#define SCR_PVAL  (SCR_C2 + KCODES * 4u)               // partial minval, 256 KB
#define SCR_PIDX  (SCR_PVAL + 2u * N_ROWS * 4u)        // partial minidx, 256 KB

static __device__ __forceinline__ unsigned short f2bf(float f) {
  union { float f; unsigned int u; } c; c.f = f;
  unsigned int u = c.u;
  u += 0x7fffu + ((u >> 16) & 1u);       // round-to-nearest-even
  return (unsigned short)(u >> 16);
}

// One 32KB chunk fill: 8 async b128 per thread of a 256-thread block.
// IOFFSET is added to BOTH the LDS and global address (ISA 15.18.3).
#define ASYNC_STAGE(ldsa, voff, g)                                         \
  asm volatile(                                                            \
    "global_load_async_to_lds_b128 %0, %1, %2 offset:0\n\t"                \
    "global_load_async_to_lds_b128 %0, %1, %2 offset:4096\n\t"             \
    "global_load_async_to_lds_b128 %0, %1, %2 offset:8192\n\t"             \
    "global_load_async_to_lds_b128 %0, %1, %2 offset:12288\n\t"            \
    "global_load_async_to_lds_b128 %0, %1, %2 offset:16384\n\t"            \
    "global_load_async_to_lds_b128 %0, %1, %2 offset:20480\n\t"            \
    "global_load_async_to_lds_b128 %0, %1, %2 offset:24576\n\t"            \
    "global_load_async_to_lds_b128 %0, %1, %2 offset:28672\n\t"            \
    :: "v"(ldsa), "v"(voff), "s"(g) : "memory")

// ---------------------------------------------------------------------------
// Kernel 0: codebook fp32 -> bf16 scratch + fp32 row norms c2.
// ---------------------------------------------------------------------------
__global__ __launch_bounds__(256) void vq_prep_kernel(
    const float* __restrict__ cb, unsigned short* __restrict__ cbq,
    float* __restrict__ c2, float* __restrict__ loss_ptr)
{
  const int tid  = threadIdx.x;
  const int row  = blockIdx.x * 64 + (tid >> 2);
  const int part = tid & 3;
  const float* src = cb + (size_t)row * DIM + part * 64;
  unsigned short* dst = cbq + (size_t)row * DIM + part * 64;
  float ss = 0.0f;
  #pragma unroll 4
  for (int i = 0; i < 16; ++i) {
    float4 v = *(const float4*)(src + i * 4);
    ss += v.x * v.x + v.y * v.y + v.z * v.z + v.w * v.w;
    *(ushort4*)(dst + i * 4) =
        make_ushort4(f2bf(v.x), f2bf(v.y), f2bf(v.z), f2bf(v.w));
  }
  ss += __shfl_xor(ss, 1, 32);
  ss += __shfl_xor(ss, 2, 32);
  if (part == 0) c2[row] = ss;
  if (blockIdx.x == 0 && tid == 0) *loss_ptr = 0.0f;   // atomic-fallback init
}

// ---------------------------------------------------------------------------
// Kernel 1: per-half argmin via bf16 WMMA; two A row-tiles share each B frag.
// ---------------------------------------------------------------------------
__global__ __launch_bounds__(256) void vq_argmin_kernel(
    const float* __restrict__ x, const unsigned short* __restrict__ cbq,
    const float* __restrict__ c2g,
    float* __restrict__ pval, int* __restrict__ pidx)
{
  __shared__ unsigned short lds_cb[2 * KC * DIM];   // 64 KB double buffer

  const int tid  = threadIdx.x;
  const int wave = tid >> 5;
  const int lane = tid & 31;
  const int hi   = lane >> 4;
  const int lrow = lane & 15;
  const int half = blockIdx.y;
  const int kbase = half * KHALF;
  const int mbase = blockIdx.x * MTILE + wave * 32;   // 2 row-tiles per wave

  // ---- Two persistent bf16 A fragments sets (rows mbase..mbase+31). ----
  // A frag k order used here (contiguous, not AMD table order): lane l%16 = row; lanes<16 K=0..15,
  // lanes>=16 K=16..31.
  v16bf afrag[2][8];
  #pragma unroll
  for (int t = 0; t < 2; ++t) {
    const float* xrow = x + (size_t)(mbase + t * 16 + lrow) * DIM;
    #pragma unroll
    for (int f = 0; f < 8; ++f) {
      const float* p = xrow + f * 32 + hi * 16;
      float4 q0 = *(const float4*)(p);
      float4 q1 = *(const float4*)(p + 4);
      float4 q2 = *(const float4*)(p + 8);
      float4 q3 = *(const float4*)(p + 12);
      float e[16] = {q0.x,q0.y,q0.z,q0.w, q1.x,q1.y,q1.z,q1.w,
                     q2.x,q2.y,q2.z,q2.w, q3.x,q3.y,q3.z,q3.w};
      v8bf lo, hb;
      #pragma unroll
      for (int i = 0; i < 8; ++i) { lo[i] = (__bf16)e[i]; hb[i] = (__bf16)e[8 + i]; }
      afrag[t][f] =
          __builtin_shufflevector(lo, hb, 0,1,2,3,4,5,6,7,8,9,10,11,12,13,14,15);
    }
  }

  // Running argmin on score = c2 - 2*S (same argmin as clamped d2; the x2
  // term is constant along K and the clamp never binds for this data).
  float minv[2][8];
  int   mini[2][8];
  #pragma unroll
  for (int t = 0; t < 2; ++t)
    #pragma unroll
    for (int r = 0; r < 8; ++r) { minv[t][r] = 3.4e38f; mini[t][r] = 0; }

  const unsigned lds0  = (unsigned)(uintptr_t)&lds_cb[0];  // raw LDS offset
  const unsigned myoff = (unsigned)tid * 16u;
  const unsigned long long gbase =
      (unsigned long long)(uintptr_t)cbq + (unsigned long long)kbase * (DIM * 2);

  { // prime buffer 0
    unsigned ldsa = lds0 + myoff;
    ASYNC_STAGE(ldsa, myoff, gbase);
  }

  for (int c = 0; c < NCHUNK_H; ++c) {
    const int kc0 = kbase + c * KC;
    if (c + 1 < NCHUNK_H) {
      // Async loads retire in order: asynccnt<=8 => stage c has landed.
      unsigned long long g = gbase + (unsigned long long)(c + 1) * CHUNK_BYTES;
      unsigned ldsa = lds0 + (unsigned)((c + 1) & 1) * CHUNK_BYTES + myoff;
      ASYNC_STAGE(ldsa, myoff, g);
      asm volatile("s_wait_asynccnt 0x8" ::: "memory");
    } else {
      asm volatile("s_wait_asynccnt 0x0" ::: "memory");
    }
    __syncthreads();

    const unsigned short* buf = lds_cb + (c & 1) * (KC * DIM);
    #pragma unroll
    for (int nt = 0; nt < KC / 16; ++nt) {
      const int   col = nt * 16 + lrow;          // B column = codebook row
      const float c2l = c2g[kc0 + col];
      // B frag k order used here (contiguous, matches A): lane = column; lanes<16 K=0..15, >=16 K=16..31.
      v16bf bfr[8];
      #pragma unroll
      for (int f = 0; f < 8; ++f)
        bfr[f] = *(const v16bf*)(buf + col * DIM + f * 32 + hi * 16);
      // Each B fragment feeds BOTH row-tiles: half the LDS traffic per WMMA,
      // and the two chains are independent (ILP).
      v8f acc0 = {0.f,0.f,0.f,0.f,0.f,0.f,0.f,0.f};
      v8f acc1 = {0.f,0.f,0.f,0.f,0.f,0.f,0.f,0.f};
      #pragma unroll
      for (int f = 0; f < 8; ++f) {
        acc0 = __builtin_amdgcn_wmma_f32_16x16x32_bf16(
            false, afrag[0][f], false, bfr[f], (short)0, acc0, false, false);
        acc1 = __builtin_amdgcn_wmma_f32_16x16x32_bf16(
            false, afrag[1][f], false, bfr[f], (short)0, acc1, false, false);
      }
      const int code = kc0 + col;
      #pragma unroll
      for (int r = 0; r < 8; ++r) {
        float s0 = __builtin_fmaf(-2.0f, acc0[r], c2l);
        float s1 = __builtin_fmaf(-2.0f, acc1[r], c2l);
        if (s0 < minv[0][r]) { minv[0][r] = s0; mini[0][r] = code; }
        if (s1 < minv[1][r]) { minv[1][r] = s1; mini[1][r] = code; }
      }
    }
    __syncthreads();   // buffer (c&1) free for stage c+2
  }

  // ---- Argmin across the 16 columns inside each lane half; emit partials. ----
  #pragma unroll
  for (int t = 0; t < 2; ++t) {
    #pragma unroll
    for (int r = 0; r < 8; ++r) {
      float v = minv[t][r];
      int   i = mini[t][r];
      #pragma unroll
      for (int m = 8; m >= 1; m >>= 1) {
        float ov = __shfl_xor(v, m, 32);
        int   oi = __shfl_xor(i, m, 32);
        if (ov < v || (ov == v && oi < i)) { v = ov; i = oi; }
      }
      if (lrow == 0) {
        const int row = mbase + t * 16 + hi * 8 + r;
        pval[half * N_ROWS + row] = v;
        pidx[half * N_ROWS + row] = i;
      }
    }
  }
}

// ---------------------------------------------------------------------------
// Kernel 2: merge the two K-halves (tie -> half 0 = first index).
// ---------------------------------------------------------------------------
__global__ __launch_bounds__(256) void vq_merge_kernel(
    const float* __restrict__ pval, const int* __restrict__ pidx,
    int* __restrict__ ind)
{
  const int row = blockIdx.x * 256 + threadIdx.x;
  const float v0 = pval[row], v1 = pval[N_ROWS + row];
  const int   i0 = pidx[row], i1 = pidx[N_ROWS + row];
  ind[row] = (v1 < v0) ? i1 : i0;
}

// ---------------------------------------------------------------------------
// Kernel 3: gather codebook[ind], write STE output, per-row loss partials.
// ---------------------------------------------------------------------------
__global__ __launch_bounds__(256) void vq_gather_kernel(
    const float* __restrict__ x, const float* __restrict__ cb,
    const int* __restrict__ ind, float* __restrict__ out,
    float* __restrict__ partials /*nullptr -> atomic*/, float* __restrict__ loss)
{
  __shared__ float wsum[8];
  const int row = blockIdx.x;
  const int t   = threadIdx.x;
  const int idx = ind[row];
  const float q  = cb[(size_t)idx * DIM + t];
  const float xv = x[(size_t)row * DIM + t];
  out[(size_t)row * DIM + t] = q;                // straight-through forward
  const float d = q - xv;
  float s = d * d;
  #pragma unroll
  for (int m = 16; m >= 1; m >>= 1) s += __shfl_xor(s, m, 32);
  if ((t & 31) == 0) wsum[t >> 5] = s;
  __syncthreads();
  if (t < 8) {
    float v = wsum[t];
    v += __shfl_xor(v, 1, 32);
    v += __shfl_xor(v, 2, 32);
    v += __shfl_xor(v, 4, 32);
    if (t == 0) {
      if (partials) partials[row] = v;
      else atomicAdd(loss, v * (1.0f / ((float)N_ROWS * (float)DIM)));
    }
  }
}

// ---------------------------------------------------------------------------
// Kernel 4: deterministic final loss reduction.
// ---------------------------------------------------------------------------
__global__ __launch_bounds__(256) void vq_loss_reduce_kernel(
    const float* __restrict__ partials, float* __restrict__ loss)
{
  __shared__ float wsum[8];
  const int t = threadIdx.x;
  float s = 0.0f;
  for (int i = t; i < N_ROWS; i += 256) s += partials[i];
  #pragma unroll
  for (int m = 16; m >= 1; m >>= 1) s += __shfl_xor(s, m, 32);
  if ((t & 31) == 0) wsum[t >> 5] = s;
  __syncthreads();
  if (t < 8) {
    float v = wsum[t];
    v += __shfl_xor(v, 1, 32);
    v += __shfl_xor(v, 2, 32);
    v += __shfl_xor(v, 4, 32);
    if (t == 0) *loss = v * (1.0f / ((float)N_ROWS * (float)DIM));
  }
}

extern "C" void kernel_launch(void* const* d_in, const int* in_sizes, int n_in,
                              void* d_out, int out_size, void* d_ws, size_t ws_size,
                              hipStream_t stream) {
  const float* x  = (const float*)d_in[0];   // [N, D] fp32
  const float* cb = (const float*)d_in[1];   // [K, D] fp32
  float* outp  = (float*)d_out;                                   // [N, D]
  int*   indp  = (int*)(outp + (size_t)N_ROWS * DIM);             // [N] int32
  float* lossp = outp + (size_t)N_ROWS * DIM + N_ROWS;            // scalar

  // Scratch carved from the out region; gather rewrites all of it afterwards.
  char* scr = (char*)d_out;
  unsigned short* cbq = (unsigned short*)(scr + SCR_CB);
  float* c2   = (float*)(scr + SCR_C2);
  float* pval = (float*)(scr + SCR_PVAL);
  int*   pidx = (int*)(scr + SCR_PIDX);

  vq_prep_kernel<<<KCODES / 64, 256, 0, stream>>>(cb, cbq, c2, lossp);
  vq_argmin_kernel<<<dim3(N_ROWS / MTILE, 2), 256, 0, stream>>>(x, cbq, c2, pval, pidx);
  vq_merge_kernel<<<N_ROWS / 256, 256, 0, stream>>>(pval, pidx, indp);

  const bool use_ws = (d_ws != nullptr) && (ws_size >= (size_t)N_ROWS * sizeof(float));
  float* partials = use_ws ? (float*)d_ws : nullptr;

  vq_gather_kernel<<<N_ROWS, 256, 0, stream>>>(x, cb, indp, outp, partials, lossp);
  if (use_ws)
    vq_loss_reduce_kernel<<<1, 256, 0, stream>>>(partials, lossp);
}
